// RWKV7_1563368095782
// MI455X (gfx1250) — matrix operand order repaired and checked by reading, not compiled, not run on hardware
//
#include <hip/hip_runtime.h>
#include <hip/hip_bf16.h>
#include <cstdint>

// ---------------- problem constants ----------------
constexpr int Bz = 8, Tz = 2048, Cz = 768;
constexpr int Hz = 12, Nh = 64;
constexpr int Mrows = Bz * Tz;           // 16384
constexpr float EPS_GN = 0.00064f;

// ---------------- WMMA / TDM types ----------------
typedef __attribute__((ext_vector_type(16))) __bf16 v16bf;
typedef __attribute__((ext_vector_type(8)))  float  v8f;
typedef __attribute__((ext_vector_type(4)))  unsigned int u32x4;
typedef __attribute__((ext_vector_type(8)))  int i32x8;
typedef __attribute__((ext_vector_type(4)))  int i32x4;

__device__ __forceinline__ __bf16 f2bf(float f) {
    unsigned u = __float_as_uint(f);
    u += 0x7FFFu + ((u >> 16) & 1u);          // round-to-nearest-even
    unsigned short h = (unsigned short)(u >> 16);
    return __builtin_bit_cast(__bf16, h);
}
__device__ __forceinline__ unsigned pack2bf(float lo, float hi) {
    unsigned a = __float_as_uint(lo); a += 0x7FFFu + ((a >> 16) & 1u);
    unsigned b = __float_as_uint(hi); b += 0x7FFFu + ((b >> 16) & 1u);
    return (a >> 16) | (b & 0xFFFF0000u);
}
__device__ __forceinline__ float sigmoidf_(float x) { return 1.0f / (1.0f + __expf(-x)); }

// =====================================================================
// K0: token shift + 4-way mixing (time_maa): produces xrg,xwa,xk,xv
// =====================================================================
__global__ __launch_bounds__(128) void mix_kernel(
    const float* __restrict__ x,
    const float* __restrict__ tmx, const float* __restrict__ tmrg,
    const float* __restrict__ tmwa, const float* __restrict__ tmk,
    const float* __restrict__ tmv,
    const float* __restrict__ w1,   // [C,112]
    const float* __restrict__ w2,   // [4,28,C]
    float* __restrict__ xrg, float* __restrict__ xwa,
    float* __restrict__ xk,  float* __restrict__ xv)
{
    int row = blockIdx.x;
    int t   = row % Tz;
    __shared__ float sx[Cz], sxx[Cz], sxm[Cz], sh[112];
    const float* xp = x + (size_t)row * Cz;
    for (int c = threadIdx.x; c < Cz; c += 128) {
        float xc  = xp[c];
        float xxc = (t == 0 ? 0.0f : xp[c - Cz]) - xc;
        sx[c] = xc; sxx[c] = xxc; sxm[c] = xc + xxc * tmx[c];
    }
    __syncthreads();
    for (int u = threadIdx.x; u < 112; u += 128) {
        float acc = 0.0f;
        for (int c = 0; c < Cz; ++c) acc += sxm[c] * w1[c * 112 + u];
        sh[u] = tanhf(acc);
    }
    __syncthreads();
    for (int c = threadIdx.x; c < Cz; c += 128) {
        float m0 = 0, m1 = 0, m2 = 0, m3 = 0;
        for (int d = 0; d < 28; ++d) {
            m0 += sh[d]      * w2[(0 * 28 + d) * Cz + c];
            m1 += sh[28 + d] * w2[(1 * 28 + d) * Cz + c];
            m2 += sh[56 + d] * w2[(2 * 28 + d) * Cz + c];
            m3 += sh[84 + d] * w2[(3 * 28 + d) * Cz + c];
        }
        float xc = sx[c], xxc = sxx[c];
        size_t o = (size_t)row * Cz + c;
        xrg[o] = xc + xxc * (tmrg[c] + m0);
        xwa[o] = xc + xxc * (tmwa[c] + m1);
        xk[o]  = xc + xxc * (tmk[c]  + m2);
        xv[o]  = xc + xxc * (tmv[c]  + m3);
    }
}

// =====================================================================
// K1: bf16 WMMA GEMM  Y[M,N] = X[M,K] @ W[K,N]   (f32 in/out)
// 256 threads / 8 wave32, tile 128x128, k-step 32, double-buffered LDS:
// global float4 loads for tile k+1 issue before the WMMA block of tile k.
// =====================================================================
constexpr int BM = 128, BN = 128, BK = 32;
constexpr int LDA = BK + 8;
constexpr int LDB = BK + 8;

__global__ __launch_bounds__(256) void gemm_bf16_wmma(
    const float* __restrict__ X, const float* __restrict__ W,
    float* __restrict__ Y, int Ktot, int Ntot)
{
    __shared__ alignas(16) __bf16 As[2][BM * LDA];
    __shared__ alignas(16) __bf16 Bs[2][BN * LDB];

    const int bm = blockIdx.x * BM;
    const int bn = blockIdx.y * BN;
    const int tid  = threadIdx.x;
    const int wave = tid >> 5;
    const int lane = tid & 31;
    const int kg   = lane >> 4;
    const int lm   = lane & 15;
    const int wm   = (wave >> 1) * 32;
    const int wn   = (wave & 1) * 64;

    v8f zero = {0, 0, 0, 0, 0, 0, 0, 0};
    v8f acc[2][4];
    #pragma unroll
    for (int i = 0; i < 2; ++i)
        #pragma unroll
        for (int j = 0; j < 4; ++j) acc[i][j] = zero;

    float4 fa[4], fb[4];

    auto load_tile = [&](int k0) {
        #pragma unroll
        for (int l = 0; l < 4; ++l) {
            int idx = tid + l * 256;
            int rr = idx >> 3, c4 = (idx & 7) << 2;
            fa[l] = *(const float4*)&X[(size_t)(bm + rr) * Ktot + k0 + c4];
        }
        #pragma unroll
        for (int l = 0; l < 4; ++l) {
            int idx = tid + l * 256;
            int kk = idx >> 5, n4 = (idx & 31) << 2;
            fb[l] = *(const float4*)&W[(size_t)(k0 + kk) * Ntot + bn + n4];
        }
    };
    auto store_tile = [&](int buf) {
        #pragma unroll
        for (int l = 0; l < 4; ++l) {
            int idx = tid + l * 256;
            int rr = idx >> 3, c4 = (idx & 7) << 2;
            uint2 p; p.x = pack2bf(fa[l].x, fa[l].y); p.y = pack2bf(fa[l].z, fa[l].w);
            *(uint2*)&As[buf][rr * LDA + c4] = p;
        }
        #pragma unroll
        for (int l = 0; l < 4; ++l) {
            int idx = tid + l * 256;
            int kk = idx >> 5, n4 = (idx & 31) << 2;
            Bs[buf][(n4 + 0) * LDB + kk] = f2bf(fb[l].x);
            Bs[buf][(n4 + 1) * LDB + kk] = f2bf(fb[l].y);
            Bs[buf][(n4 + 2) * LDB + kk] = f2bf(fb[l].z);
            Bs[buf][(n4 + 3) * LDB + kk] = f2bf(fb[l].w);
        }
    };

    load_tile(0);
    store_tile(0);
    __syncthreads();

    int cur = 0;
    for (int k0 = 0; k0 < Ktot; k0 += BK) {
        bool more = (k0 + BK) < Ktot;
        if (more) load_tile(k0 + BK);   // global loads fly during WMMA below

        v16bf afrag[2], bfrag[4];
        #pragma unroll
        for (int rt = 0; rt < 2; ++rt) {
            int rowi = wm + rt * 16 + lm;
            union { float4 q[2]; v16bf v; } ua;
            ua.q[0] = *(const float4*)&As[cur][rowi * LDA + 8 * kg];
            ua.q[1] = *(const float4*)&As[cur][rowi * LDA + 16 + 8 * kg];
            afrag[rt] = ua.v;
        }
        #pragma unroll
        for (int ct = 0; ct < 4; ++ct) {
            int coli = wn + ct * 16 + lm;
            union { float4 q[2]; v16bf v; } ub;
            ub.q[0] = *(const float4*)&Bs[cur][coli * LDB + 8 * kg];
            ub.q[1] = *(const float4*)&Bs[cur][coli * LDB + 16 + 8 * kg];
            bfrag[ct] = ub.v;
        }
        #pragma unroll
        for (int rt = 0; rt < 2; ++rt)
            #pragma unroll
            for (int ct = 0; ct < 4; ++ct)
                acc[rt][ct] = __builtin_amdgcn_wmma_f32_16x16x32_bf16(
                    false, afrag[rt], false, bfrag[ct],
                    (short)0, acc[rt][ct], false, false);

        if (more) {
            store_tile(cur ^ 1);        // writes other buffer: no conflict
            __syncthreads();            // one barrier per k-iteration
            cur ^= 1;
        }
    }

    #pragma unroll
    for (int rt = 0; rt < 2; ++rt)
        #pragma unroll
        for (int ct = 0; ct < 4; ++ct) {
            int col = bn + wn + ct * 16 + lm;
            #pragma unroll
            for (int p = 0; p < 8; ++p) {
                int rowo = bm + wm + rt * 16 + p + 8 * kg;
                Y[(size_t)rowo * Ntot + col] = acc[rt][ct][p];
            }
        }
}

// =====================================================================
// K2: bottleneck projections + elementwise fusions (one block per row)
// =====================================================================
__global__ __launch_bounds__(128) void proj_kernel(
    const float* __restrict__ xrg, const float* __restrict__ xwa,
    const float* __restrict__ xk,  const float* __restrict__ xv,
    const float* __restrict__ k0g, const float* __restrict__ v0g,
    const float* __restrict__ v1g,
    const float* __restrict__ td,   const float* __restrict__ dw1, const float* __restrict__ dw2,
    const float* __restrict__ aaa1, const float* __restrict__ aaa2, const float* __restrict__ taaaaa,
    const float* __restrict__ kkk1, const float* __restrict__ kkk2,
    const float* __restrict__ g1,   const float* __restrict__ g2,
    const float* __restrict__ ma1,  const float* __restrict__ ma2, const float* __restrict__ tma,
    const float* __restrict__ mk1,  const float* __restrict__ mk2, const float* __restrict__ tmk_,
    const float* __restrict__ mv1,  const float* __restrict__ mv2, const float* __restrict__ tmv_,
    float* __restrict__ wdec, float* __restrict__ kf, float* __restrict__ vf,
    float* __restrict__ aaout, float* __restrict__ bbout, float* __restrict__ gout)
{
    int row = blockIdx.x;
    size_t o0 = (size_t)row * Cz;
    __shared__ float srg[Cz], swa[Cz], skx[Cz], svx[Cz], sk0[Cz], sv0[Cz], sv1[Cz];
    __shared__ float skk[Cz];
    __shared__ float hd[64], ha[16], hma[16], hmk[16], hmv[16], hkkk[16], hg[120];
    __shared__ float snorm[Hz];

    for (int c = threadIdx.x; c < Cz; c += 128) {
        srg[c] = xrg[o0 + c]; swa[c] = xwa[o0 + c];
        skx[c] = xk[o0 + c];  svx[c] = xv[o0 + c];
        sk0[c] = k0g[o0 + c]; sv0[c] = v0g[o0 + c]; sv1[c] = v1g[o0 + c];
    }
    __syncthreads();

    for (int u = threadIdx.x; u < 264; u += 128) {
        float acc = 0.0f;
        if (u < 64) {
            for (int c = 0; c < Cz; ++c) acc += swa[c] * dw1[c * 64 + u];
            hd[u] = tanhf(acc);
        } else if (u < 80) {
            int j = u - 64;
            for (int c = 0; c < Cz; ++c) acc += swa[c] * aaa1[c * 16 + j];
            ha[j] = acc;
        } else if (u < 96) {
            int j = u - 80;
            for (int c = 0; c < Cz; ++c) acc += swa[c] * ma1[c * 16 + j];
            hma[j] = acc;
        } else if (u < 112) {
            int j = u - 96;
            for (int c = 0; c < Cz; ++c) acc += skx[c] * mk1[c * 16 + j];
            hmk[j] = acc;
        } else if (u < 128) {
            int j = u - 112;
            for (int c = 0; c < Cz; ++c) acc += svx[c] * mv1[c * 16 + j];
            hmv[j] = acc;
        } else if (u < 144) {
            int j = u - 128;
            for (int c = 0; c < Cz; ++c) acc += skx[c] * kkk1[c * 16 + j];
            hkkk[j] = tanhf(acc);
        } else {
            int j = u - 144;
            for (int c = 0; c < Cz; ++c) acc += srg[c] * g1[c * 120 + j];
            hg[j] = sigmoidf_(acc);
        }
    }
    __syncthreads();

    float wd_l[6], a_l[6], ma_l[6], mk_l[6], mv_l[6], g_l[6];
    int idx = 0;
    for (int c = threadIdx.x; c < Cz; c += 128, ++idx) {
        float zd = td[c];
        for (int j = 0; j < 64; ++j) zd += hd[j] * dw2[j * Cz + c];
        wd_l[idx] = -log1pf(__expf(-zd)) - 0.5f;

        float av = taaaaa[c], mav = tma[c], mkv = tmk_[c], mvv = tmv_[c], kkd = 0.0f;
        for (int j = 0; j < 16; ++j) {
            av  += ha[j]   * aaa2[j * Cz + c];
            mav += hma[j]  * ma2[j * Cz + c];
            mkv += hmk[j]  * mk2[j * Cz + c];
            mvv += hmv[j]  * mv2[j * Cz + c];
            kkd += hkkk[j] * kkk2[j * Cz + c];
        }
        a_l[idx]  = sigmoidf_(av);
        ma_l[idx] = sigmoidf_(mav);
        mk_l[idx] = sigmoidf_(mkv);
        mv_l[idx] = sigmoidf_(mvv);
        float gv = 0.0f;
        for (int j = 0; j < 120; ++j) gv += hg[j] * g2[j * Cz + c];
        g_l[idx] = gv;
        skk[c] = sk0[c] + kkd;
    }
    __syncthreads();
    if (threadIdx.x < Hz) {
        int h = threadIdx.x; float ss = 0.0f;
        for (int i = 0; i < Nh; ++i) { float v = skk[h * Nh + i]; ss += v * v; }
        snorm[h] = 1.0f / fmaxf(sqrtf(ss), 1e-12f);
    }
    __syncthreads();
    idx = 0;
    for (int c = threadIdx.x; c < Cz; c += 128, ++idx) {
        float kkn = skk[c] * snorm[c / Nh];
        float a   = a_l[idx], mav = ma_l[idx];
        float kv  = sk0[c];
        float k   = kv * mav + kv * a * (1.0f - mav);
        k *= __expf(fminf(wd_l[idx] * mk_l[idx], 0.0f));
        float vv  = sv0[c] + (sv1[c] - sv0[c]) * mv_l[idx];
        size_t o = o0 + c;
        wdec[o]  = wd_l[idx];
        kf[o]    = k;
        vf[o]    = vv;
        aaout[o] = -kkn;
        bbout[o] = kkn * a;
        gout[o]  = g_l[idx];
    }
}

// =====================================================================
// K3: RWKV7 state scan with TDM (Tensor Data Mover) double buffering.
// One block per (b,h). Thread i owns state row i in 64 registers.
// TDM fetches 16-timestep x 64-float tiles of r/w/k/v/aa/bb (stride C)
// into LDS for chunk c+1 while VALU processes chunk c (TENSORcnt sync).
// NOTE: single __shared__ object => its LDS offset is 0, so the D#
// lds_addr is just the arithmetic offset into sbuf.
// =====================================================================
constexpr int CT = 16;               // timesteps per chunk
constexpr int NCHUNK = Tz / CT;      // 128

#if __has_builtin(__builtin_amdgcn_tensor_load_to_lds)
#define USE_TDM 1
#else
#define USE_TDM 0
#endif

#if USE_TDM
__device__ __forceinline__ void tdm_issue_tile(const float* gsrc, unsigned lds_byte_off) {
    // D# group0: count=1, lds_addr, global_addr, type=2 ("image")
    unsigned long long ga = (unsigned long long)(uintptr_t)gsrc;
    u32x4 g0;
    g0[0] = 1u;
    g0[1] = lds_byte_off;
    g0[2] = (unsigned)ga;
    g0[3] = (unsigned)((ga >> 32) & 0x1FFFFFFull) | (2u << 30);
    // D# group1: data_size=4B(code 2); 2-D tile 64 x CT; tensor dims = tile dims;
    // tensor_dim0_stride = Cz (row stride in elements)
    i32x8 g1;
    g1[0] = (2 << 16);                       // workgroup_mask=0 | data_size=2
    g1[1] = (int)(64u << 16);                // tensor_dim0[15:0] in [31:16]
    g1[2] = (int)(((unsigned)CT) << 16);     // tensor_dim0[31:16]=0 | tensor_dim1[15:0]
    g1[3] = (int)(64u << 16);                // tensor_dim1[31:16]=0 | tile_dim0=64
    g1[4] = CT;                              // tile_dim1=CT | tile_dim2=0
    g1[5] = Cz;                              // tensor_dim0_stride low 32
    g1[6] = 0;                               // stride hi | dim1_stride lo
    g1[7] = 0;
    i32x4 gz = {0, 0, 0, 0};                 // groups 2/3 unused (2-D tensor)
#if defined(__clang_major__) && __clang_major__ >= 23
    i32x8 g4 = {0, 0, 0, 0, 0, 0, 0, 0};
    __builtin_amdgcn_tensor_load_to_lds(g0, g1, gz, gz, g4, 0);
#else
    __builtin_amdgcn_tensor_load_to_lds(g0, g1, gz, gz, 0);
#endif
}
#endif

__global__ __launch_bounds__(64) void scan_kernel(
    const float* __restrict__ r,  const float* __restrict__ wraw,
    const float* __restrict__ k,  const float* __restrict__ v,
    const float* __restrict__ aa, const float* __restrict__ bb,
    const float* __restrict__ state, const float* __restrict__ jrt_alpha,
    float* __restrict__ y, float* __restrict__ s_out)
{
    // ONLY shared object in this kernel (LDS offset 0 assumed for TDM).
    __shared__ float sbuf[2][6][CT * Nh];    // 48 KB

    int bh = blockIdx.x;
    int b = bh / Hz, h = bh % Hz;
    int i = threadIdx.x;
    float alpha = sigmoidf_(jrt_alpha[0]);

    float s[Nh], inj[Nh];
    const float* st = state + ((size_t)bh) * Nh * Nh + (size_t)i * Nh;
    #pragma unroll
    for (int j = 0; j < Nh; ++j) { float sv = st[j]; s[j] = sv; inj[j] = sv * alpha; }

    const float* srcs[6] = { r, wraw, k, v, aa, bb };
    size_t base = (size_t)b * Tz * Cz + (size_t)h * Nh;

#if USE_TDM
    // prime chunk 0
    if (threadIdx.x < 32) {
        #pragma unroll
        for (int a = 0; a < 6; ++a)
            tdm_issue_tile(srcs[a] + base, (unsigned)((0 * 6 + a) * CT * Nh * 4));
#if __has_builtin(__builtin_amdgcn_s_wait_tensorcnt)
        __builtin_amdgcn_s_wait_tensorcnt((short)0);
#endif
    }
    __syncthreads();
    for (int e = threadIdx.x; e < CT * Nh; e += 64)
        sbuf[0][1][e] = __expf(-__expf(sbuf[0][1][e]));   // w -> exp(-exp(w))
    __syncthreads();
#endif

    #pragma unroll 1
    for (int c = 0; c < NCHUNK; ++c) {
        int cur = c & 1, nb = cur ^ 1;
#if USE_TDM
        if (c + 1 < NCHUNK && threadIdx.x < 32) {
            size_t coff = base + (size_t)(c + 1) * CT * Cz;
            #pragma unroll
            for (int a = 0; a < 6; ++a)
                tdm_issue_tile(srcs[a] + coff, (unsigned)((nb * 6 + a) * CT * Nh * 4));
        }
#else
        // fallback: cooperative synchronous load of chunk c
        {
            size_t coff = base + (size_t)c * CT * Cz;
            for (int a = 0; a < 6; ++a)
                for (int e = threadIdx.x; e < CT * Nh; e += 64)
                    sbuf[cur][a][e] = srcs[a][coff + (size_t)(e >> 6) * Cz + (e & 63)];
        }
        __syncthreads();
        for (int e = threadIdx.x; e < CT * Nh; e += 64)
            sbuf[cur][1][e] = __expf(-__expf(sbuf[cur][1][e]));
        __syncthreads();
#endif
        // ---- compute CT steps from resident chunk (no per-step barriers) ----
        const float* lr = &sbuf[cur][0][0];
        const float* lw = &sbuf[cur][1][0];
        const float* lk = &sbuf[cur][2][0];
        const float* lv = &sbuf[cur][3][0];
        const float* la = &sbuf[cur][4][0];
        const float* lb = &sbuf[cur][5][0];
        size_t yo = base + (size_t)c * CT * Cz + i;
        #pragma unroll 1
        for (int tl = 0; tl < CT; ++tl) {
            int off = tl * Nh;
            float sa = 0.0f;
            #pragma unroll
            for (int j = 0; j < Nh; ++j) sa += s[j] * la[off + j];
            float vi = lv[off + i];
            float yi = 0.0f;
            #pragma unroll
            for (int j = 0; j < Nh; ++j) {
                float sj = s[j] * lw[off + j] + sa * lb[off + j] + vi * lk[off + j] + inj[j];
                s[j] = sj;
                yi += sj * lr[off + j];
            }
            y[yo + (size_t)tl * Cz] = yi;
        }
#if USE_TDM
        if (c + 1 < NCHUNK) {
            if (threadIdx.x < 32) {
#if __has_builtin(__builtin_amdgcn_s_wait_tensorcnt)
                __builtin_amdgcn_s_wait_tensorcnt((short)0);
#endif
            }
            __syncthreads();
            for (int e = threadIdx.x; e < CT * Nh; e += 64)
                sbuf[nb][1][e] = __expf(-__expf(sbuf[nb][1][e]));
            __syncthreads();
        }
#endif
    }
    float* so = s_out + ((size_t)bh) * Nh * Nh + (size_t)i * Nh;
    #pragma unroll
    for (int j = 0; j < Nh; ++j) so[j] = s[j];
}

// =====================================================================
// K4: GroupNorm(H groups, eps=0.00064, biased) + bonus + gate -> z
// =====================================================================
__global__ __launch_bounds__(128) void post_kernel(
    const float* __restrict__ y, const float* __restrict__ r,
    const float* __restrict__ k, const float* __restrict__ v,
    const float* __restrict__ g,
    const float* __restrict__ lnw, const float* __restrict__ lnb,
    const float* __restrict__ faaaa, float* __restrict__ z)
{
    int row = blockIdx.x;
    size_t o0 = (size_t)row * Cz;
    __shared__ float sy[Cz], sr[Cz], sk2[Cz], sv2[Cz];
    __shared__ float mu[Hz], rinv[Hz], bon[Hz];
    for (int c = threadIdx.x; c < Cz; c += 128) {
        sy[c] = y[o0 + c]; sr[c] = r[o0 + c];
        sk2[c] = k[o0 + c]; sv2[c] = v[o0 + c];
    }
    __syncthreads();
    if (threadIdx.x < Hz) {
        int h = threadIdx.x;
        float m = 0.0f;
        for (int i = 0; i < Nh; ++i) m += sy[h * Nh + i];
        m /= (float)Nh;
        float vv = 0.0f;
        for (int i = 0; i < Nh; ++i) { float d = sy[h * Nh + i] - m; vv += d * d; }
        vv /= (float)Nh;
        mu[h] = m; rinv[h] = rsqrtf(vv + EPS_GN);
        float bn = 0.0f;
        for (int i = 0; i < Nh; ++i) bn += sr[h * Nh + i] * sk2[h * Nh + i] * faaaa[h * Nh + i];
        bon[h] = bn;
    }
    __syncthreads();
    for (int c = threadIdx.x; c < Cz; c += 128) {
        int h = c / Nh;
        float yn = (sy[c] - mu[h]) * rinv[h] * lnw[c] + lnb[c];
        z[o0 + c] = (yn + bon[h] * sv2[c]) * g[o0 + c];
    }
}

// =====================================================================
extern "C" void kernel_launch(void* const* d_in, const int* in_sizes, int n_in,
                              void* d_out, int out_size, void* d_ws, size_t ws_size,
                              hipStream_t stream)
{
    const float* x       = (const float*)d_in[0];
    const float* v1      = (const float*)d_in[1];
    const float* state   = (const float*)d_in[2];
    const float* tmx     = (const float*)d_in[3];
    const float* tmrg    = (const float*)d_in[4];
    const float* tmwa    = (const float*)d_in[5];
    const float* tmk     = (const float*)d_in[6];
    const float* tmv     = (const float*)d_in[7];
    const float* maa_w1  = (const float*)d_in[8];
    const float* maa_w2  = (const float*)d_in[9];
    const float* tdec    = (const float*)d_in[10];
    const float* dw1     = (const float*)d_in[11];
    const float* dw2     = (const float*)d_in[12];
    const float* aaa1    = (const float*)d_in[13];
    const float* aaa2    = (const float*)d_in[14];
    const float* taaaaa  = (const float*)d_in[15];
    const float* kkk1    = (const float*)d_in[16];
    const float* kkk2    = (const float*)d_in[17];
    const float* g1      = (const float*)d_in[18];
    const float* g2      = (const float*)d_in[19];
    const float* ma1     = (const float*)d_in[20];
    const float* ma2     = (const float*)d_in[21];
    const float* tma     = (const float*)d_in[22];
    const float* mk1     = (const float*)d_in[23];
    const float* mk2     = (const float*)d_in[24];
    const float* tmk_    = (const float*)d_in[25];
    const float* mv1     = (const float*)d_in[26];
    const float* mv2     = (const float*)d_in[27];
    const float* tmv_    = (const float*)d_in[28];
    const float* faaaa   = (const float*)d_in[29];
    const float* Wr      = (const float*)d_in[30];
    const float* Wk      = (const float*)d_in[31];
    const float* Wv      = (const float*)d_in[32];
    const float* Wo      = (const float*)d_in[33];
    const float* lnw     = (const float*)d_in[34];
    const float* lnb     = (const float*)d_in[35];
    const float* jalpha  = (const float*)d_in[36];

    float* out = (float*)d_out;
    const size_t S = (size_t)Mrows * Cz;

    float* ws  = (float*)d_ws;
    float* xrg = ws + 0 * S;   // later reused as z
    float* xwa = ws + 1 * S;   // later reused as wd
    float* xk  = ws + 2 * S;   // later reused as aa
    float* xv  = ws + 3 * S;   // later reused as bb
    float* rb  = ws + 4 * S;
    float* kf  = ws + 5 * S;
    float* vf  = ws + 6 * S;
    float* gb  = ws + 7 * S;
    float* yb  = ws + 8 * S;
    float* zb  = xrg;
    float* wd  = xwa;
    float* aab = xk;
    float* bbb = xv;

    float* xout  = out;
    float* v1out = out + S;
    float* sout  = out + 2 * S;

    mix_kernel<<<Mrows, 128, 0, stream>>>(x, tmx, tmrg, tmwa, tmk, tmv,
                                          maa_w1, maa_w2, xrg, xwa, xk, xv);

    dim3 ggrid(Mrows / BM, Cz / BN);
    gemm_bf16_wmma<<<ggrid, 256, 0, stream>>>(xrg, Wr, rb, Cz, Cz);
    gemm_bf16_wmma<<<ggrid, 256, 0, stream>>>(xk,  Wk, kf, Cz, Cz);
    gemm_bf16_wmma<<<ggrid, 256, 0, stream>>>(xv,  Wv, vf, Cz, Cz);

    proj_kernel<<<Mrows, 128, 0, stream>>>(
        xrg, xwa, xk, xv, kf, vf, v1,
        tdec, dw1, dw2, aaa1, aaa2, taaaaa, kkk1, kkk2,
        g1, g2, ma1, ma2, tma, mk1, mk2, tmk_, mv1, mv2, tmv_,
        wd, kf, vf, aab, bbb, gb);

    scan_kernel<<<Bz * Hz, 64, 0, stream>>>(rb, wd, kf, vf, aab, bbb,
                                            state, jalpha, yb, sout);

    post_kernel<<<Mrows, 128, 0, stream>>>(yb, rb, kf, vf, gb,
                                           lnw, lnb, faaaa, zb);

    gemm_bf16_wmma<<<ggrid, 256, 0, stream>>>(zb, Wo, xout, Cz, Cz);

    hipMemcpyAsync(v1out, v1, S * sizeof(float), hipMemcpyDeviceToDevice, stream);
}
